// LinearAttention2_6012954214508
// MI455X (gfx1250) — matrix operand order repaired and checked by reading, not compiled, not run on hardware
//
#include <hip/hip_runtime.h>
#include <hip/hip_bf16.h>

// ---------------------------------------------------------------------------
// Linear attention, CDNA5 (gfx1250) WMMA implementation.
// Big GEMMs: LDS-staged, double-buffered with GLOBAL_LOAD_ASYNC_TO_LDS_B128
// (ASYNCcnt) overlapping v_wmma_f32_16x16x32_f16 compute.
// b=4, n=4096, c=768, h=12, d=64. Outputs: out [b,n,c] f32, kv [b,h,d,e] f32.
// ---------------------------------------------------------------------------

typedef _Float16 v8h  __attribute__((ext_vector_type(8)));
typedef _Float16 v16h __attribute__((ext_vector_type(16)));
typedef float    v8f  __attribute__((ext_vector_type(8)));
typedef int      v4i  __attribute__((vector_size(16)));   // matches builtin param

#define B_   4
#define N_   4096
#define C_   768
#define H_   12
#define D_   64
#define CK_  (3 * C_)          // 2304
#define MTOT (B_ * N_)         // 16384
#define BH_  (B_ * H_)         // 48

#define APITCH 40              // LDS row pitch in halves (32 data + 8 pad) -> 20 dwords,
                               // bank-conflict-free for 16B ds reads across 16 lanes

// ---- async global->LDS copy (16 bytes per lane) ---------------------------
#if __has_builtin(__builtin_amdgcn_global_load_async_to_lds_b128)
#define COPY16(g, s)                                                          \
  __builtin_amdgcn_global_load_async_to_lds_b128(                             \
      (__attribute__((address_space(1))) v4i*)(uintptr_t)(g),                 \
      (__attribute__((address_space(3))) v4i*)(unsigned)(uintptr_t)(s), 0, 0)
#else
__device__ __forceinline__ void async_copy16_asm(const _Float16* g, _Float16* s) {
  unsigned lds_off = (unsigned)(uintptr_t)s;
  asm volatile("global_load_async_to_lds_b128 %0, %1, off"
               :: "v"(lds_off), "v"(g) : "memory");
}
#define COPY16(g, s) async_copy16_asm((g), (s))
#endif

#if __has_builtin(__builtin_amdgcn_s_wait_asynccnt)
#define ASYNC_WAIT() __builtin_amdgcn_s_wait_asynccnt(0)
#else
#define ASYNC_WAIT() asm volatile("s_wait_asynccnt 0" ::: "memory")
#endif

__device__ __forceinline__ v8f wmma_f16(v16h a, v16h b, v8f c) {
  return __builtin_amdgcn_wmma_f32_16x16x32_f16(false, a, false, b, (short)0, c,
                                                false, false);
}

__device__ __forceinline__ v16h pack16(v8h lo, v8h hh) {
  v16h a;
#pragma unroll
  for (int i = 0; i < 8; ++i) { a[i] = lo[i]; a[8 + i] = hh[i]; }
  return a;
}

// ---- fragment loads straight from global (row-major, for small GEMMs) -----
__device__ __forceinline__ v16h load_a_frag(const _Float16* __restrict__ base,
                                            int lda, int m0, int k0, int l, int hi) {
  const _Float16* p = base + (size_t)(m0 + l) * lda + (k0 + hi * 8);
  return pack16(*(const v8h*)p, *(const v8h*)(p + 16));
}
__device__ __forceinline__ v16h load_b_frag(const _Float16* __restrict__ bt,
                                            int ldb, int n0, int k0, int l, int hi) {
  const _Float16* p = bt + (size_t)(n0 + l) * ldb + (k0 + hi * 8);
  return pack16(*(const v8h*)p, *(const v8h*)(p + 16));
}

// ---- fragment loads from LDS-staged 128x32 tiles (pitch APITCH) -----------
__device__ __forceinline__ v16h lds_a_frag(const _Float16* ab, int m_local,
                                           int l, int hi) {
  const _Float16* p = ab + (m_local + l) * APITCH + hi * 8;
  return pack16(*(const v8h*)p, *(const v8h*)(p + 16));
}
__device__ __forceinline__ v16h lds_b_frag(const _Float16* bb, int n_local,
                                           int l, int hi) {
  const _Float16* p = bb + (n_local + l) * APITCH + hi * 8;
  return pack16(*(const v8h*)p, *(const v8h*)(p + 16));
}

// Copy one 128(M)x32(K) A slice + 128(N)x32(K) B slice into an LDS buffer.
// 256 threads x 2 iterations x (16B A + 16B B) = full tile; 4 async ops/thread.
__device__ __forceinline__ void copy_tile(_Float16* buf,
                                          const _Float16* __restrict__ Ag, int lda, int m0,
                                          const _Float16* __restrict__ Bg, int ldb, int n0,
                                          int k0, int tid) {
#pragma unroll
  for (int j = 0; j < 2; ++j) {
    const int idx = (tid << 1) | j;            // 0..511
    const int row = idx >> 2;                  // 0..127
    const int c = (idx & 3) << 3;              // 0,8,16,24 (halves)
    COPY16(Ag + (size_t)(m0 + row) * lda + k0 + c, buf + row * APITCH + c);
    COPY16(Bg + (size_t)(n0 + row) * ldb + k0 + c, buf + (128 + row) * APITCH + c);
  }
}

// ---------------------------------------------------------------------------
// f32 -> f16 conversion (grid-stride)
// ---------------------------------------------------------------------------
__global__ void cvt_f16_kernel(const float* __restrict__ src,
                               _Float16* __restrict__ dst, int n) {
  for (int i = blockIdx.x * blockDim.x + threadIdx.x; i < n;
       i += gridDim.x * blockDim.x)
    dst[i] = (_Float16)src[i];
}

// ---------------------------------------------------------------------------
// Stage 1: qkv = x @ w_qkv^T  (M=16384, N=2304, K=768), LDS double-buffered.
// Epilogue: elu+1 on q,k; q -> [b,h,s,d], k -> [b,h,d,s], v -> [b,h,e,s] (f16)
// ---------------------------------------------------------------------------
__global__ __launch_bounds__(256) void qkv_gemm_kernel(
    const _Float16* __restrict__ xh, const _Float16* __restrict__ wh,
    _Float16* __restrict__ qh, _Float16* __restrict__ kTh,
    _Float16* __restrict__ vTh) {
  __shared__ __align__(16) _Float16 lds[2][256 * APITCH];  // 40 KB

  const int tid = threadIdx.x;
  const int lane = tid & 31, wave = tid >> 5;
  const int l = lane & 15, hi = lane >> 4;
  const int m0 = blockIdx.x * 128, n0 = blockIdx.y * 128;
  const int mw = (wave >> 1) * 32, nw = (wave & 1) * 64;

  v8f acc[2][4];
#pragma unroll
  for (int mi = 0; mi < 2; ++mi)
#pragma unroll
    for (int ni = 0; ni < 4; ++ni) acc[mi][ni] = (v8f){};

  copy_tile(&lds[0][0], xh, C_, m0, wh, C_, n0, 0, tid);
  for (int kt = 0; kt < C_ / 32; ++kt) {
    ASYNC_WAIT();
    __syncthreads();
    if (kt + 1 < C_ / 32)
      copy_tile(&lds[(kt + 1) & 1][0], xh, C_, m0, wh, C_, n0, (kt + 1) * 32, tid);

    const _Float16* ab = &lds[kt & 1][0];
    const _Float16* bb = ab + 128 * APITCH;
    v16h a[2], bfr[4];
#pragma unroll
    for (int mi = 0; mi < 2; ++mi) a[mi] = lds_a_frag(ab, mw + mi * 16, l, hi);
#pragma unroll
    for (int ni = 0; ni < 4; ++ni) bfr[ni] = lds_b_frag(bb, nw + ni * 16, l, hi);
#pragma unroll
    for (int mi = 0; mi < 2; ++mi)
#pragma unroll
      for (int ni = 0; ni < 4; ++ni)
        acc[mi][ni] = wmma_f16(a[mi], bfr[ni], acc[mi][ni]);
  }

#pragma unroll
  for (int mi = 0; mi < 2; ++mi)
#pragma unroll
    for (int ni = 0; ni < 4; ++ni)
#pragma unroll
      for (int i = 0; i < 8; ++i) {
        const int m = m0 + mw + mi * 16 + hi * 8 + i;
        const int n = n0 + nw + ni * 16 + l;
        const float v = acc[mi][ni][i];
        const int which = n / C_;  // 0=q, 1=k, 2=v
        const int rem = n % C_;
        const int h = rem >> 6, dc = rem & 63;
        const int b = m >> 12, s = m & (N_ - 1);
        const size_t bh = (size_t)(b * H_ + h);
        if (which == 0) {
          const float f = (v > 0.f) ? (v + 1.f) : __expf(v);  // elu(v)+1
          qh[(bh * N_ + s) * D_ + dc] = (_Float16)f;
        } else if (which == 1) {
          const float f = (v > 0.f) ? (v + 1.f) : __expf(v);
          kTh[(bh * D_ + dc) * N_ + s] = (_Float16)f;
        } else {
          vTh[(bh * D_ + dc) * N_ + s] = (_Float16)v;
        }
      }
}

// ---------------------------------------------------------------------------
// Stage 2: ksum[bh,d] = sum_s k[bh,d,s]
// ---------------------------------------------------------------------------
__global__ __launch_bounds__(256) void ksum_kernel(
    const _Float16* __restrict__ kTh, float* __restrict__ ksum) {
  const int row = blockIdx.x;  // bh*64 + d
  const _Float16* p = kTh + (size_t)row * N_;
  float s = 0.f;
  for (int i = threadIdx.x; i < N_; i += 256) s += (float)p[i];
  __shared__ float red[256];
  red[threadIdx.x] = s;
  __syncthreads();
  for (int off = 128; off > 0; off >>= 1) {
    if (threadIdx.x < off) red[threadIdx.x] += red[threadIdx.x + off];
    __syncthreads();
  }
  if (threadIdx.x == 0) ksum[row] = red[0];
}

// ---------------------------------------------------------------------------
// Stage 3: z[bh,s] = 1 / (q[bh,s,:] . ksum[bh,:])  (f32; value can exceed f16 max)
// ---------------------------------------------------------------------------
__global__ __launch_bounds__(256) void z_kernel(const _Float16* __restrict__ qh,
                                                const float* __restrict__ ksum,
                                                float* __restrict__ z) {
  const int idx = blockIdx.x * 256 + threadIdx.x;  // over BH*N
  const int bh = idx >> 12, s = idx & (N_ - 1);
  const _Float16* q = qh + ((size_t)bh * N_ + s) * D_;
  const float* ks = ksum + (size_t)bh * D_;
  float acc = 0.f;
#pragma unroll 8
  for (int d = 0; d < D_; ++d) acc += (float)q[d] * ks[d];
  z[idx] = 1.0f / acc;
}

// ---------------------------------------------------------------------------
// Stage 4: kv^T[e,d] = sum_s v[s,e] k[s,d] per (b,h). 8 waves x s-chunks of 512,
// cross-wave reduce via LDS ds_add_f32. Writes kv f32 (output #2) + kv^T f16.
// ---------------------------------------------------------------------------
__global__ __launch_bounds__(256) void kv_kernel(
    const _Float16* __restrict__ vTh, const _Float16* __restrict__ kTh,
    _Float16* __restrict__ kvTh, float* __restrict__ kv_out) {
  const int bh = blockIdx.x;
  const _Float16* A = vTh + (size_t)bh * D_ * N_;
  const _Float16* Bt = kTh + (size_t)bh * D_ * N_;

  __shared__ float buf[D_ * D_];
  for (int i = threadIdx.x; i < D_ * D_; i += 256) buf[i] = 0.f;
  __syncthreads();

  const int lane = threadIdx.x & 31, wave = threadIdx.x >> 5;
  const int l = lane & 15, hi = lane >> 4;
  const int s_base = wave * 512;

  v8f acc[4][4];
#pragma unroll
  for (int mi = 0; mi < 4; ++mi)
#pragma unroll
    for (int ni = 0; ni < 4; ++ni) acc[mi][ni] = (v8f){};

  for (int k0 = s_base; k0 < s_base + 512; k0 += 32) {
    v16h a[4], bb[4];
#pragma unroll
    for (int mi = 0; mi < 4; ++mi) a[mi] = load_a_frag(A, N_, mi * 16, k0, l, hi);
#pragma unroll
    for (int ni = 0; ni < 4; ++ni) bb[ni] = load_b_frag(Bt, N_, ni * 16, k0, l, hi);
#pragma unroll
    for (int mi = 0; mi < 4; ++mi)
#pragma unroll
      for (int ni = 0; ni < 4; ++ni)
        acc[mi][ni] = wmma_f16(a[mi], bb[ni], acc[mi][ni]);
  }

#pragma unroll
  for (int mi = 0; mi < 4; ++mi)
#pragma unroll
    for (int ni = 0; ni < 4; ++ni)
#pragma unroll
      for (int i = 0; i < 8; ++i) {
        const int e = mi * 16 + hi * 8 + i;
        const int d = ni * 16 + l;
        atomicAdd(&buf[e * D_ + d], acc[mi][ni][i]);
      }
  __syncthreads();

  for (int i = threadIdx.x; i < D_ * D_; i += 256) {
    const int e = i >> 6, d = i & 63;
    const float v = buf[i];
    kvTh[(size_t)bh * D_ * D_ + i] = (_Float16)v;       // [e,d]
    kv_out[(size_t)bh * D_ * D_ + d * D_ + e] = v;      // [d,e]
  }
}

// ---------------------------------------------------------------------------
// Stage 5: out_attn[s,e] = z[s] * sum_d q[s,d] kv[d,e] per (b,h).
// ---------------------------------------------------------------------------
__global__ __launch_bounds__(128) void attn_kernel(
    const _Float16* __restrict__ qh, const _Float16* __restrict__ kvTh,
    const float* __restrict__ z, _Float16* __restrict__ attnh) {
  const int bh = blockIdx.y;
  const int b = bh / H_, h = bh % H_;
  const int lane = threadIdx.x & 31, wave = threadIdx.x >> 5;
  const int l = lane & 15, hi = lane >> 4;
  const int m_base = blockIdx.x * 256 + wave * 64;

  const _Float16* A = qh + (size_t)bh * N_ * D_;
  const _Float16* Bt = kvTh + (size_t)bh * D_ * D_;

  v8f acc[4][4];
#pragma unroll
  for (int mi = 0; mi < 4; ++mi)
#pragma unroll
    for (int ni = 0; ni < 4; ++ni) acc[mi][ni] = (v8f){};

  for (int k0 = 0; k0 < D_; k0 += 32) {
    v16h a[4], bb[4];
#pragma unroll
    for (int mi = 0; mi < 4; ++mi)
      a[mi] = load_a_frag(A, D_, m_base + mi * 16, k0, l, hi);
#pragma unroll
    for (int ni = 0; ni < 4; ++ni)
      bb[ni] = load_b_frag(Bt, D_, ni * 16, k0, l, hi);
#pragma unroll
    for (int mi = 0; mi < 4; ++mi)
#pragma unroll
      for (int ni = 0; ni < 4; ++ni)
        acc[mi][ni] = wmma_f16(a[mi], bb[ni], acc[mi][ni]);
  }

  const float* zp = z + (size_t)bh * N_;
#pragma unroll
  for (int mi = 0; mi < 4; ++mi)
#pragma unroll
    for (int ni = 0; ni < 4; ++ni)
#pragma unroll
      for (int i = 0; i < 8; ++i) {
        const int s = m_base + mi * 16 + hi * 8 + i;
        const int e = ni * 16 + l;
        const float v = acc[mi][ni][i] * zp[s];
        attnh[((size_t)b * N_ + s) * C_ + h * D_ + e] = (_Float16)v;
      }
}

// ---------------------------------------------------------------------------
// Stage 6: out = attn @ w_proj^T + b_proj  (M=16384, N=768, K=768),
// LDS double-buffered with async copies, f32 out.
// ---------------------------------------------------------------------------
__global__ __launch_bounds__(256) void proj_gemm_kernel(
    const _Float16* __restrict__ attnh, const _Float16* __restrict__ wph,
    const float* __restrict__ bproj, float* __restrict__ out) {
  __shared__ __align__(16) _Float16 lds[2][256 * APITCH];  // 40 KB

  const int tid = threadIdx.x;
  const int lane = tid & 31, wave = tid >> 5;
  const int l = lane & 15, hi = lane >> 4;
  const int m0 = blockIdx.x * 128, n0 = blockIdx.y * 128;
  const int mw = (wave >> 1) * 32, nw = (wave & 1) * 64;

  v8f acc[2][4];
#pragma unroll
  for (int mi = 0; mi < 2; ++mi)
#pragma unroll
    for (int ni = 0; ni < 4; ++ni) acc[mi][ni] = (v8f){};

  copy_tile(&lds[0][0], attnh, C_, m0, wph, C_, n0, 0, tid);
  for (int kt = 0; kt < C_ / 32; ++kt) {
    ASYNC_WAIT();
    __syncthreads();
    if (kt + 1 < C_ / 32)
      copy_tile(&lds[(kt + 1) & 1][0], attnh, C_, m0, wph, C_, n0,
                (kt + 1) * 32, tid);

    const _Float16* ab = &lds[kt & 1][0];
    const _Float16* bb = ab + 128 * APITCH;
    v16h a[2], bfr[4];
#pragma unroll
    for (int mi = 0; mi < 2; ++mi) a[mi] = lds_a_frag(ab, mw + mi * 16, l, hi);
#pragma unroll
    for (int ni = 0; ni < 4; ++ni) bfr[ni] = lds_b_frag(bb, nw + ni * 16, l, hi);
#pragma unroll
    for (int mi = 0; mi < 2; ++mi)
#pragma unroll
      for (int ni = 0; ni < 4; ++ni)
        acc[mi][ni] = wmma_f16(a[mi], bfr[ni], acc[mi][ni]);
  }

#pragma unroll
  for (int mi = 0; mi < 2; ++mi)
#pragma unroll
    for (int ni = 0; ni < 4; ++ni)
#pragma unroll
      for (int i = 0; i < 8; ++i) {
        const int m = m0 + mw + mi * 16 + hi * 8 + i;
        const int n = n0 + nw + ni * 16 + l;
        out[(size_t)m * C_ + n] = acc[mi][ni][i] + bproj[n];
      }
}

// ---------------------------------------------------------------------------
// Host launcher
// ---------------------------------------------------------------------------
extern "C" void kernel_launch(void* const* d_in, const int* in_sizes, int n_in,
                              void* d_out, int out_size, void* d_ws,
                              size_t ws_size, hipStream_t stream) {
  const float* x      = (const float*)d_in[0];
  const float* w_qkv  = (const float*)d_in[1];
  const float* w_proj = (const float*)d_in[2];
  const float* b_proj = (const float*)d_in[3];

  float* out    = (float*)d_out;
  float* kv_out = out + (size_t)B_ * N_ * C_;

  // Workspace layout (all offsets 256B-aligned)
  char* ws = (char*)d_ws;
  size_t off = 0;
  _Float16* xh    = (_Float16*)(ws + off); off += (size_t)MTOT * C_ * 2;
  _Float16* wqh   = (_Float16*)(ws + off); off += (size_t)CK_ * C_ * 2;
  _Float16* wph   = (_Float16*)(ws + off); off += (size_t)C_ * C_ * 2;
  _Float16* qh    = (_Float16*)(ws + off); off += (size_t)MTOT * C_ * 2;
  _Float16* kTh   = (_Float16*)(ws + off); off += (size_t)MTOT * C_ * 2;
  _Float16* vTh   = (_Float16*)(ws + off); off += (size_t)MTOT * C_ * 2;
  _Float16* kvTh  = (_Float16*)(ws + off); off += (size_t)BH_ * D_ * D_ * 2;
  float*    ksum  = (float*)(ws + off);    off += (size_t)BH_ * D_ * 4;
  float*    z     = (float*)(ws + off);    off += (size_t)BH_ * N_ * 4;
  _Float16* attnh = (_Float16*)(ws + off); off += (size_t)MTOT * C_ * 2;
  (void)ws_size; (void)in_sizes; (void)n_in; (void)out_size;

  // Convert inputs to f16
  cvt_f16_kernel<<<4096, 256, 0, stream>>>(x, xh, MTOT * C_);
  cvt_f16_kernel<<<2048, 256, 0, stream>>>(w_qkv, wqh, CK_ * C_);
  cvt_f16_kernel<<<1024, 256, 0, stream>>>(w_proj, wph, C_ * C_);

  // qkv projection + elu feature maps (+ transposed k/v layouts)
  qkv_gemm_kernel<<<dim3(MTOT / 128, CK_ / 128), 256, 0, stream>>>(
      xh, wqh, qh, kTh, vTh);

  // normalizer pieces
  ksum_kernel<<<BH_ * D_, 256, 0, stream>>>(kTh, ksum);
  z_kernel<<<(BH_ * N_) / 256, 256, 0, stream>>>(qh, ksum, z);

  // kv aggregation (also output #2)
  kv_kernel<<<BH_, 256, 0, stream>>>(vTh, kTh, kvTh, kv_out);

  // attention output
  attn_kernel<<<dim3(N_ / 256, BH_), 128, 0, stream>>>(qh, kvTh, z, attnh);

  // final projection + bias
  proj_gemm_kernel<<<dim3(MTOT / 128, C_ / 128), 256, 0, stream>>>(
      attnh, wph, b_proj, out);
}
